// Net_17514876633598
// MI455X (gfx1250) — matrix operand order repaired and checked by reading, not compiled, not run on hardware
//
#include <hip/hip_runtime.h>
#include <hip/hip_bf16.h>

typedef __attribute__((ext_vector_type(16))) __bf16 v16bf;
typedef __attribute__((ext_vector_type(8)))  float  v8f;

// ---------------- utility kernels ----------------

__global__ void k_zero(float* p, long n) {
  long i = (long)blockIdx.x * blockDim.x + threadIdx.x;
  long stride = (long)gridDim.x * blockDim.x;
  for (; i < n; i += stride) p[i] = 0.0f;
}

// per-(relation,dst) edge counts (counts identical for both layers)
__global__ void k_count(const int* __restrict__ etype, const int* __restrict__ dst,
                        float* __restrict__ cnt, int E, int N) {
  int e = blockIdx.x * blockDim.x + threadIdx.x;
  if (e < E) atomicAdd(&cnt[(long)etype[e] * N + dst[e]], 1.0f);
}

__global__ void k_invcnt(float* cnt, long n) {
  long i = (long)blockIdx.x * blockDim.x + threadIdx.x;
  long stride = (long)gridDim.x * blockDim.x;
  for (; i < n; i += stride) cnt[i] = 1.0f / fmaxf(cnt[i], 1.0f);
}

// gather feat[src] rows, scatter-add into agg[dst] for edges of one relation.
// 16 threads per edge, float4 per thread (coalesced 256B row reads, L2-resident).
__global__ void k_scatter(const float* __restrict__ feat, const int* __restrict__ src,
                          const int* __restrict__ dst, const int* __restrict__ etype,
                          int rel, float* __restrict__ agg, int E) {
  int t = blockIdx.x * blockDim.x + threadIdx.x;
  int e = t >> 4;
  if (e >= E) return;
  if (etype[e] != rel) return;
  int g = t & 15;
  const float4 v = reinterpret_cast<const float4*>(feat + (long)src[e] * 64)[g];
  float* a = agg + (long)dst[e] * 64 + g * 4;
  atomicAdd(a + 0, v.x);
  atomicAdd(a + 1, v.y);
  atomicAdd(a + 2, v.z);
  atomicAdd(a + 3, v.w);
}

// pack 8 contiguous K values (two float4 runs) into fragment slots [base, base+8)
__device__ __forceinline__ void pack8(v16bf& v, int base, float4 lo, float4 hi, float s) {
  v[base + 0] = (__bf16)(lo.x * s);
  v[base + 1] = (__bf16)(lo.y * s);
  v[base + 2] = (__bf16)(lo.z * s);
  v[base + 3] = (__bf16)(lo.w * s);
  v[base + 4] = (__bf16)(hi.x * s);
  v[base + 5] = (__bf16)(hi.y * s);
  v[base + 6] = (__bf16)(hi.z * s);
  v[base + 7] = (__bf16)(hi.w * s);
}

// ---------------- WMMA GEMM: OUT[N x 64] (+)= (rowscale . X[N x 64]) @ W[64 x 64] (+ bias) ----
// One wave computes a 16x16 C tile; 8 waves/block -> 32 rows x 64 cols per block.
// Fragment layouts per CDNA5 ISA 7.12.2 (16-bit A 16x32, B 32x16, 32-bit C 16x16).
// flags: bit0 = accumulate into OUT (read C from OUT), bit1 = apply ReLU at store.
// Requires Nn % 32 == 0 (holds: 100000); guard is wave-uniform so EXEC stays all-ones.
__global__ void k_gemm64(const float* __restrict__ X, const float* __restrict__ W,
                         const float* __restrict__ bias, const float* __restrict__ rowscale,
                         float* __restrict__ OUT, int Nn, int flags) {
  const int lane = threadIdx.x & 31;
  const int wave = threadIdx.x >> 5;
  const int row0 = (blockIdx.x * 2 + (wave >> 2)) * 16;  // M tile base
  const int colb = (wave & 3) * 16;                      // N tile base
  if (row0 + 16 > Nn) return;                            // uniform per wave
  const int n    = lane & 15;   // column within tile / M row within A tile
  const int half = lane >> 4;   // K-half selector

  const int row = row0 + n;
  const float s = rowscale ? rowscale[row] : 1.0f;
  const float* xr = X + (size_t)row * 64 + half * 8;

  // A fragments: element i -> K = (i/8)*16 + half*8 + (i%8); contiguous 32B runs.
  float4 q0 = *(const float4*)(xr + 0);
  float4 q1 = *(const float4*)(xr + 4);
  float4 q2 = *(const float4*)(xr + 16);
  float4 q3 = *(const float4*)(xr + 20);
  float4 q4 = *(const float4*)(xr + 32);
  float4 q5 = *(const float4*)(xr + 36);
  float4 q6 = *(const float4*)(xr + 48);
  float4 q7 = *(const float4*)(xr + 52);

  v16bf a0, a1, b0, b1;
  pack8(a0, 0, q0, q1, s);
  pack8(a0, 8, q2, q3, s);
  pack8(a1, 0, q4, q5, s);
  pack8(a1, 8, q6, q7, s);

  // B (32x16 bf16): element i -> K = (i/8)*16 + half*8 + (i%8) of column n.
  const float* wp = W + (size_t)(half * 8) * 64 + colb + n;
#pragma unroll
  for (int i = 0; i < 16; ++i) {
    b0[i] = (__bf16)wp[(i + (i & 8)) * 64];
    b1[i] = (__bf16)wp[(32 + i + (i & 8)) * 64];
  }

  // C/D tile base: VGPR r -> row (row0 + r + half*8), col (colb + n)
  float* op = OUT + (size_t)(row0 + half * 8) * 64 + colb + n;
  v8f c;
  if (flags & 1) {
#pragma unroll
    for (int r = 0; r < 8; ++r) c[r] = op[r * 64];
  } else {
    const float bv = bias[colb + n];
#pragma unroll
    for (int r = 0; r < 8; ++r) c[r] = bv;
  }

  c = __builtin_amdgcn_wmma_f32_16x16x32_bf16(false, a0, false, b0, (short)0, c, false, false);
  c = __builtin_amdgcn_wmma_f32_16x16x32_bf16(false, a1, false, b1, (short)0, c, false, false);

  if (flags & 2) {
#pragma unroll
    for (int r = 0; r < 8; ++r) c[r] = fmaxf(c[r], 0.0f);
  }
#pragma unroll
  for (int r = 0; r < 8; ++r) op[r * 64] = c[r];
}

// out64[c] += sum_i M[i][c] * rowscale[i]  (column reduction, coalesced 256B rows)
__global__ void k_colreduce(const float* __restrict__ M, const float* __restrict__ rowscale,
                            float* __restrict__ out64, int Nn) {
  int c = threadIdx.x & 63;
  int stream0 = blockIdx.x * (blockDim.x >> 6) + (threadIdx.x >> 6);
  int nstreams = gridDim.x * (blockDim.x >> 6);
  float acc = 0.0f;
  for (int i = stream0; i < Nn; i += nstreams) {
    float s = rowscale ? rowscale[i] : 1.0f;
    acc += M[(long)i * 64 + c] * s;
  }
  atomicAdd(&out64[c], acc);
}

// g = invN*pooledH @ root2 + b2 + sum_r invN*pooledAgg_r @ W2_r ; out = log_softmax(g)
__global__ void k_final(const float* __restrict__ pooledH, const float* __restrict__ pooledAgg,
                        const float* __restrict__ root2, const float* __restrict__ b2,
                        const float* __restrict__ W2, float* __restrict__ out, float invN) {
  __shared__ float g[16];
  int j = threadIdx.x;
  if (j < 16) {
    float acc = b2[j];
    for (int k = 0; k < 64; ++k) acc += invN * pooledH[k] * root2[k * 16 + j];
    for (int r = 0; r < 8; ++r) {
      const float* Wr = W2 + r * 64 * 16;
      const float* pr = pooledAgg + r * 64;
      for (int k = 0; k < 64; ++k) acc += invN * pr[k] * Wr[k * 16 + j];
    }
    g[j] = acc;
  }
  __syncthreads();
  if (j < 16) {
    float mx = -1e30f;
    for (int t = 0; t < 16; ++t) mx = fmaxf(mx, g[t]);
    float se = 0.0f;
    for (int t = 0; t < 16; ++t) se += __expf(g[t] - mx);
    out[j] = g[j] - mx - __logf(se);
  }
}

// ---------------- host orchestration ----------------

extern "C" void kernel_launch(void* const* d_in, const int* in_sizes, int n_in,
                              void* d_out, int out_size, void* d_ws, size_t ws_size,
                              hipStream_t stream) {
  const float* x     = (const float*)d_in[0];
  const int*   ei    = (const int*)d_in[1];   // [2, E] flat: row0 = src, row1 = dst
  const int*   et    = (const int*)d_in[2];   // [E]
  const float* W1    = (const float*)d_in[3]; // [8,64,64]
  const float* root1 = (const float*)d_in[4]; // [64,64]
  const float* b1    = (const float*)d_in[5]; // [64]
  const float* W2    = (const float*)d_in[6]; // [8,64,16]
  const float* root2 = (const float*)d_in[7]; // [64,16]
  const float* b2    = (const float*)d_in[8]; // [16]
  float* out = (float*)d_out;

  const int E  = in_sizes[2];
  const int Nn = in_sizes[0] / 64;
  const int* src = ei;
  const int* dst = ei + E;

  // workspace layout (floats): ~55 MB total
  float* ws        = (float*)d_ws;
  float* cnt       = ws;                          // 8*Nn  (becomes 1/max(cnt,1))
  float* agg       = cnt + (size_t)8 * Nn;        // Nn*64 (reused per relation)
  float* h         = agg + (size_t)Nn * 64;       // Nn*64 (layer-1 out, relu'd in place)
  float* pooledH   = h + (size_t)Nn * 64;         // 64
  float* pooledAgg = pooledH + 64;                // 8*64

  const dim3 B(256);
  auto nb = [](long n, int bs) { return dim3((unsigned)((n + bs - 1) / bs)); };
  const int gemmGrid = (Nn + 31) / 32;

  // counts (shared by both layers), inverted once
  k_zero<<<nb(8L * Nn, 256), B, 0, stream>>>(cnt, 8L * Nn);
  k_count<<<nb(E, 256), B, 0, stream>>>(et, dst, cnt, E, Nn);
  k_invcnt<<<nb(8L * Nn, 256), B, 0, stream>>>(cnt, 8L * Nn);
  k_zero<<<1, B, 0, stream>>>(pooledH, 64 + 8 * 64);

  // ---- layer 1: h = relu(x@root1 + b1 + sum_r (agg_r/cnt_r)@W1_r) ----
  k_gemm64<<<gemmGrid, B, 0, stream>>>(x, root1, b1, nullptr, h, Nn, 0);
  for (int r = 0; r < 8; ++r) {
    k_zero<<<nb((long)Nn * 64, 256), B, 0, stream>>>(agg, (long)Nn * 64);
    k_scatter<<<nb((long)E * 16, 256), B, 0, stream>>>(x, src, dst, et, r, agg, E);
    // last relation fuses the ReLU into the store (flags bit1)
    k_gemm64<<<gemmGrid, B, 0, stream>>>(agg, W1 + (size_t)r * 64 * 64, nullptr,
                                         cnt + (size_t)r * Nn, h, Nn, (r == 7) ? 3 : 1);
  }

  // ---- layer 2 fused with mean pool: only column means needed ----
  k_colreduce<<<512, B, 0, stream>>>(h, nullptr, pooledH, Nn);
  for (int r = 0; r < 8; ++r) {
    k_zero<<<nb((long)Nn * 64, 256), B, 0, stream>>>(agg, (long)Nn * 64);
    k_scatter<<<nb((long)E * 16, 256), B, 0, stream>>>(h, src, dst, et, r, agg, E);
    k_colreduce<<<512, B, 0, stream>>>(agg, cnt + (size_t)r * Nn, pooledAgg + r * 64, Nn);
  }

  k_final<<<1, 32, 0, stream>>>(pooledH, pooledAgg, root2, b2, W2, out, 1.0f / (float)Nn);
}
